// MambaEncoder_32804960207129
// MI455X (gfx1250) — matrix operand order repaired and checked by reading, not compiled, not run on hardware
//
#include <hip/hip_runtime.h>
#include <hip/hip_bf16.h>
#include <math.h>

// ---------------- problem constants ----------------
#define BB   2
#define SS   1024
#define DD   512
#define DH   256
#define DI   512
#define NST  16
#define KCV  4
#define LL   2
#define MM   (BB*SS)   // 2048
#define DBLW 64        // padded width of the x-proj output (48 -> 64)
#define LROW 72        // LDS row pitch in halves (64 + 8 pad, banks conflict-free)

typedef _Float16 v16h __attribute__((ext_vector_type(16)));
typedef _Float16 v8h  __attribute__((ext_vector_type(8)));
typedef float    v8f  __attribute__((ext_vector_type(8)));
typedef float    v4f  __attribute__((ext_vector_type(4)));

// ---------------- weight transpose + f32->f16 convert ----------------
// W [nrows][kcols] f32 -> Wt [kpad][nstride] f16, zero pad k>=kcols, n>=nrows
__global__ void k_wtrans(const float* __restrict__ W, _Float16* __restrict__ Wt,
                         int nrows, int nstride, int kcols, int kpad) {
    int idx = blockIdx.x * blockDim.x + threadIdx.x;
    int total = kpad * nstride;
    if (idx >= total) return;
    int k = idx / nstride, n = idx - k * nstride;
    float v = (k < kcols && n < nrows) ? W[(size_t)n * kcols + k] : 0.f;
    Wt[idx] = (_Float16)v;
}

// ---------------- slice copy with optional time flip ----------------
__global__ void k_slice(const float* __restrict__ src, float* __restrict__ dst,
                        int srcStride, int srcOff, int dstStride, int dstOff,
                        int cols, int flip) {
    int idx = blockIdx.x * blockDim.x + threadIdx.x;
    if (idx >= MM * cols) return;
    int m = idx / cols, c = idx - m * cols;
    int b = m / SS, t = m - b * SS;
    int td = flip ? (SS - 1 - t) : t;
    dst[((size_t)(b * SS + td)) * dstStride + dstOff + c] =
        src[((size_t)(b * SS + t)) * srcStride + srcOff + c];
}

// ---------------- layernorm: one wave32 per row ----------------
__global__ void k_ln(const float* __restrict__ in, const float* __restrict__ g,
                     const float* __restrict__ bta, float* __restrict__ out, int width) {
    int row  = blockIdx.x * (blockDim.x >> 5) + (threadIdx.x >> 5);
    int lane = threadIdx.x & 31;
    if (row >= MM) return;
    const float* p = in + (size_t)row * width;
    float s = 0.f;
    for (int c = lane; c < width; c += 32) s += p[c];
    #pragma unroll
    for (int o = 16; o; o >>= 1) s += __shfl_xor(s, o, 32);
    float mean = s / width;
    float v = 0.f;
    for (int c = lane; c < width; c += 32) { float d = p[c] - mean; v += d * d; }
    #pragma unroll
    for (int o = 16; o; o >>= 1) v += __shfl_xor(v, o, 32);
    float rstd = rsqrtf(v / width + 1e-5f);
    float* q = out + (size_t)row * width;
    for (int c = lane; c < width; c += 32)
        q[c] = (p[c] - mean) * rstd * g[c] + bta[c];
}

// ---------------- causal depthwise conv (K=4) + SiLU ----------------
__global__ void k_conv(const float* __restrict__ xz, const float* __restrict__ w,
                       const float* __restrict__ cb, float* __restrict__ out) {
    int idx = blockIdx.x * blockDim.x + threadIdx.x;
    if (idx >= MM * DI) return;
    int m = idx / DI, c = idx - m * DI;
    int b = m / SS, t = m - b * SS;
    float acc = cb[c];
    #pragma unroll
    for (int j = 0; j < KCV; ++j) {
        int tt = t - (KCV - 1) + j;
        if (tt >= 0)
            acc += w[c * KCV + j] * xz[((size_t)(b * SS + tt)) * 1024 + c];
    }
    out[idx] = acc / (1.f + __expf(-acc));   // silu
}

// ---------------- selective scan + D-skip + SiLU gating ----------------
__global__ __launch_bounds__(512) void k_scan(
        const float* __restrict__ dtb,  // [M][512]
        const float* __restrict__ dbl,  // [M][DBLW] cols 16..31 = B, 32..47 = C
        const float* __restrict__ xc,   // [M][512]
        const float* __restrict__ xz,   // [M][1024], z = cols 512..1023
        const float* __restrict__ Alog, // [512][16]
        const float* __restrict__ Dp,   // [512]
        float* __restrict__ out) {      // [M][512]
    __shared__ float sBC[2][32];
    int b = blockIdx.x;
    int d = threadIdx.x;
    float A[NST], h[NST];
    #pragma unroll
    for (int n = 0; n < NST; ++n) { A[n] = -__expf(Alog[d * NST + n]); h[n] = 0.f; }
    float Dv = Dp[d];
    for (int t = 0; t < SS; ++t) {
        int buf = t & 1;
        size_t m = (size_t)b * SS + t;
        if (threadIdx.x < 32) sBC[buf][threadIdx.x] = dbl[m * DBLW + 16 + threadIdx.x];
        __syncthreads();
        float dtv = dtb[m * 512 + d];
        float xv  = xc [m * 512 + d];
        float acc = 0.f;
        #pragma unroll
        for (int n = 0; n < NST; ++n) {
            h[n] = __expf(dtv * A[n]) * h[n] + dtv * sBC[buf][n] * xv;
            acc += h[n] * sBC[buf][16 + n];
        }
        float y  = acc + xv * Dv;
        float zv = xz[m * 1024 + 512 + d];
        out[m * 512 + d] = y * (zv / (1.f + __expf(-zv)));
        // next iter writes the other LDS buffer -> one barrier per step
    }
}

// ---------------- WMMA GEMM: C = epi(A @ Bt [+ bias]) [+U] ----------------
// Preconditions: Mrows%64==0, Ncols%64==0, Kdim%32==0.
// A f32 [Mrows][lda], Bt f16 [Kdim][Ncols] (K-major).
// 128 threads = 4 waves; block computes 64 rows x 64 cols.
// B tile (32K x 64N f16) staged through double-buffered LDS, fetched one
// K-step ahead (software pipeline) so WMMAs wait on DScnt, not global loads.
// Template params give straight-line epilogues (no per-element branches).
template<int MODE, bool HASB, bool HASU>
__global__ __launch_bounds__(128) void k_gemm(
        const float* __restrict__ A, int lda,
        const _Float16* __restrict__ Bt,
        const float* __restrict__ bias,
        float* __restrict__ C, int ldc,
        const float* __restrict__ U, int ldu,
        int Ncols, int Kdim) {
    __shared__ _Float16 sB[2][32 * LROW];     // 2 x 4608 B

    int tid  = threadIdx.x;
    int wave = tid >> 5;
    int lane = tid & 31;
    int mBase = (blockIdx.y * 4 + wave) * 16;
    int nBase = blockIdx.x * 64;
    int half = lane >> 4, ml = lane & 15;

    // cooperative B-tile fetch: thread -> (k row, 16-col chunk)
    int kk   = tid >> 2;
    int ccol = (tid & 3) * 16;
    const _Float16* gB  = Bt + (size_t)kk * Ncols + nBase + ccol;
    const float*    ap0 = A + (size_t)(mBase + ml) * lda + half * 8;

    // prologue: stage K-step 0 into buffer 0
    {
        v8h s0 = *(const v8h*)(gB);
        v8h s1 = *(const v8h*)(gB + 8);
        *(v8h*)(&sB[0][kk * LROW + ccol])     = s0;
        *(v8h*)(&sB[0][kk * LROW + ccol + 8]) = s1;
    }
    __syncthreads();

    v8f acc0 = {}, acc1 = {}, acc2 = {}, acc3 = {};
    int buf = 0;

    for (int k0 = 0; k0 < Kdim; k0 += 32) {
        bool more = (k0 + 32) < Kdim;          // uniform
        // ---- fetch next K-step's B tile into registers (hide latency) ----
        v8h n0 = {}, n1 = {};
        if (more) {
            const _Float16* gn = gB + (size_t)(k0 + 32) * Ncols;
            __builtin_prefetch(gn + 32 * (size_t)Ncols, 0, 1);  // 2 steps ahead
            n0 = *(const v8h*)(gn);
            n1 = *(const v8h*)(gn + 8);
        }
        // ---- A fragment: 16-bit A layout (lanes 0-15: K=half*8+e; +16 e>=8)
        const float* ap = ap0 + k0;
        v4f a0 = *(const v4f*)(ap);
        v4f a1 = *(const v4f*)(ap + 4);
        v4f a2 = *(const v4f*)(ap + 16);
        v4f a3 = *(const v4f*)(ap + 20);
        v16h a;
        #pragma unroll
        for (int e = 0; e < 4; ++e) {
            a[e]      = (_Float16)a0[e];
            a[4 + e]  = (_Float16)a1[e];
            a[8 + e]  = (_Float16)a2[e];
            a[12 + e] = (_Float16)a3[e];
        }
        // ---- B fragments from LDS: lane = N col ml, K=half*8+e; +16 e>=8 ----
        const _Float16* lp = &sB[buf][ml];
        v16h b0, b1, b2, b3;
        {
            #pragma unroll
            for (int e = 0; e < 16; ++e) {
                int kr = (e < 8) ? (half * 8 + e) : (16 + half * 8 + (e - 8));
                const _Float16* q = lp + kr * LROW;
                b0[e] = q[0];
                b1[e] = q[16];
                b2[e] = q[32];
                b3[e] = q[48];
            }
        }
        acc0 = __builtin_amdgcn_wmma_f32_16x16x32_f16(false, a, false, b0, (short)0, acc0, false, false);
        acc1 = __builtin_amdgcn_wmma_f32_16x16x32_f16(false, a, false, b1, (short)0, acc1, false, false);
        acc2 = __builtin_amdgcn_wmma_f32_16x16x32_f16(false, a, false, b2, (short)0, acc2, false, false);
        acc3 = __builtin_amdgcn_wmma_f32_16x16x32_f16(false, a, false, b3, (short)0, acc3, false, false);
        // ---- publish next tile to the other buffer; one barrier per step ----
        if (more) {
            buf ^= 1;
            *(v8h*)(&sB[buf][kk * LROW + ccol])     = n0;
            *(v8h*)(&sB[buf][kk * LROW + ccol + 8]) = n1;
            __syncthreads();
        }
    }

    // epilogue: C layout VGPR r -> row = mBase + half*8 + r, col = tile + ml
    v8f accs[4] = { acc0, acc1, acc2, acc3 };
    #pragma unroll
    for (int j = 0; j < 4; ++j) {
        int col = nBase + j * 16 + ml;
        float bv = 0.f;
        if constexpr (HASB) bv = bias[col];
        #pragma unroll
        for (int r = 0; r < 8; ++r) {
            int orow = mBase + half * 8 + r;
            float v = accs[j][r] + bv;
            if constexpr (MODE == 2) v = (v > 20.f) ? v : log1pf(__expf(v));
            if constexpr (HASU) v += U[(size_t)orow * ldu + col];
            C[(size_t)orow * ldc + col] = v;
        }
    }
}

// ---------------- host-side orchestration ----------------
static inline void launch_gemm(const float* A, int lda, const _Float16* Bt,
                               const float* bias, float* C, int ldc,
                               const float* U, int ldu,
                               int Mr, int Nc, int Kd, int mode, hipStream_t s) {
    dim3 g(Nc / 64, Mr / 64);
    if (mode == 2)
        k_gemm<2, true,  false><<<g, 128, 0, s>>>(A, lda, Bt, bias, C, ldc, U, ldu, Nc, Kd);
    else if (U)
        k_gemm<0, false, true ><<<g, 128, 0, s>>>(A, lda, Bt, bias, C, ldc, U, ldu, Nc, Kd);
    else if (bias)
        k_gemm<0, true,  false><<<g, 128, 0, s>>>(A, lda, Bt, bias, C, ldc, U, ldu, Nc, Kd);
    else
        k_gemm<0, false, false><<<g, 128, 0, s>>>(A, lda, Bt, bias, C, ldc, U, ldu, Nc, Kd);
}

extern "C" void kernel_launch(void* const* d_in, const int* in_sizes, int n_in,
                              void* d_out, int out_size, void* d_ws, size_t ws_size,
                              hipStream_t stream) {
    (void)in_sizes; (void)n_in; (void)out_size; (void)ws_size;
    const float* x      = (const float*)d_in[0];
    const float* ln_g   = (const float*)d_in[1];
    const float* ln_b   = (const float*)d_in[2];
    const float* inW    = (const float*)d_in[3];
    const float* convW  = (const float*)d_in[4];
    const float* convB  = (const float*)d_in[5];
    const float* xprojW = (const float*)d_in[6];
    const float* dtW    = (const float*)d_in[7];
    const float* dtB    = (const float*)d_in[8];
    const float* A_log  = (const float*)d_in[9];
    const float* Dp     = (const float*)d_in[10];
    const float* outW   = (const float*)d_in[11];
    const float* ipW    = (const float*)d_in[12];
    const float* ipB    = (const float*)d_in[13];
    const float* opW    = (const float*)d_in[14];
    const float* opB    = (const float*)d_in[15];
    const float* fln_g  = (const float*)d_in[16];
    const float* fln_b  = (const float*)d_in[17];

    // -------- workspace carve-up (floats first, then f16) --------
    float* f = (float*)d_ws;
    float* h    = f;  f += (size_t)MM * 512;
    float* xp   = f;  f += (size_t)MM * 512;
    float* cc   = f;  f += (size_t)MM * 512;
    float* u    = f;  f += (size_t)MM * 256;
    float* xn   = f;  f += (size_t)MM * 256;
    float* xz   = f;  f += (size_t)MM * 1024;
    float* xcv  = f;  f += (size_t)MM * 512;
    float* dbl  = f;  f += (size_t)MM * DBLW;
    float* dtb  = f;  f += (size_t)MM * 512;
    float* yg   = f;  f += (size_t)MM * 512;
    float* yo   = f;  f += (size_t)MM * 256;
    _Float16* hp = (_Float16*)f;
    _Float16* ipWt = hp; hp += (size_t)LL * 512 * 512;
    _Float16* opWt = hp; hp += (size_t)LL * 512 * 512;
    _Float16* iWt  = hp; hp += (size_t)LL * 2 * 256 * 1024;
    _Float16* xWt  = hp; hp += (size_t)LL * 2 * 512 * DBLW;
    _Float16* dWt  = hp; hp += (size_t)LL * 2 * 32 * 512;
    _Float16* oWt  = hp; hp += (size_t)LL * 2 * 512 * 256;

    // -------- h = x --------
    hipMemcpyAsync(h, x, (size_t)MM * 512 * sizeof(float),
                   hipMemcpyDeviceToDevice, stream);

    // -------- weight prep (transpose + f16 convert, K-major, padded) ----
    auto wt = [&](const float* W, _Float16* Wt, int nrows, int nstride,
                  int kcols, int kpad) {
        int total = kpad * nstride;
        k_wtrans<<<(total + 255) / 256, 256, 0, stream>>>(W, Wt, nrows, nstride,
                                                          kcols, kpad);
    };
    for (int l = 0; l < LL; ++l) {
        wt(ipW + (size_t)l * 512 * 512, ipWt + (size_t)l * 512 * 512, 512, 512, 512, 512);
        wt(opW + (size_t)l * 512 * 512, opWt + (size_t)l * 512 * 512, 512, 512, 512, 512);
        for (int dir = 0; dir < 2; ++dir) {
            int ld = l * 2 + dir;
            wt(inW    + (size_t)ld * 1024 * 256, iWt + (size_t)ld * 256 * 1024, 1024, 1024, 256, 256);
            wt(xprojW + (size_t)ld * 48 * 512,   xWt + (size_t)ld * 512 * DBLW, 48,   DBLW, 512, 512);
            wt(dtW    + (size_t)ld * 512 * 16,   dWt + (size_t)ld * 32 * 512,   512,  512,  16,  32);
            wt(outW   + (size_t)ld * 256 * 512,  oWt + (size_t)ld * 512 * 256,  256,  256,  512, 512);
        }
    }

    const int SL_BLK = 256;
    for (int l = 0; l < LL; ++l) {
        // xp = h @ ipW^T + ipB
        launch_gemm(h, 512, ipWt + (size_t)l * 512 * 512, ipB + (size_t)l * 512,
                    xp, 512, nullptr, 0, MM, 512, 512, 0, stream);
        for (int dir = 0; dir < 2; ++dir) {
            int ld = l * 2 + dir;
            // u = (flip?) xp[:, dir*256 : dir*256+256]
            k_slice<<<(MM * 256 + SL_BLK - 1) / SL_BLK, SL_BLK, 0, stream>>>(
                xp, u, 512, dir * 256, 256, 0, 256, dir);
            // xn = LN(u)
            k_ln<<<(MM + 3) / 4, 128, 0, stream>>>(
                u, ln_g + (size_t)ld * 256, ln_b + (size_t)ld * 256, xn, 256);
            // xz = xn @ iW^T
            launch_gemm(xn, 256, iWt + (size_t)ld * 256 * 1024, nullptr,
                        xz, 1024, nullptr, 0, MM, 1024, 256, 0, stream);
            // xcv = silu(causal_dwconv(xz[:, :512]))
            k_conv<<<(MM * DI + SL_BLK - 1) / SL_BLK, SL_BLK, 0, stream>>>(
                xz, convW + (size_t)ld * 512 * 4, convB + (size_t)ld * 512, xcv);
            // dbl = xcv @ xW^T  (N padded 48 -> 64 with zero weights)
            launch_gemm(xcv, 512, xWt + (size_t)ld * 512 * DBLW, nullptr,
                        dbl, DBLW, nullptr, 0, MM, DBLW, 512, 0, stream);
            // dtb = softplus(dbl[:, :16] @ dW^T + dtB)  (K zero-padded to 32)
            launch_gemm(dbl, DBLW, dWt + (size_t)ld * 32 * 512, dtB + (size_t)ld * 512,
                        dtb, 512, nullptr, 0, MM, 512, 32, 2, stream);
            // yg = gated SSM scan output
            k_scan<<<BB, 512, 0, stream>>>(dtb, dbl, xcv, xz,
                                           A_log + (size_t)ld * 512 * 16,
                                           Dp + (size_t)ld * 512, yg);
            // yo = u + yg @ oW^T
            launch_gemm(yg, 512, oWt + (size_t)ld * 512 * 256, nullptr,
                        yo, 256, u, 256, MM, 256, 512, 0, stream);
            // cc[:, dir*256:...] = (flip?) yo
            k_slice<<<(MM * 256 + SL_BLK - 1) / SL_BLK, SL_BLK, 0, stream>>>(
                yo, cc, 256, 0, 512, dir * 256, 256, dir);
        }
        // h = cc @ opW^T + opB
        launch_gemm(cc, 512, opWt + (size_t)l * 512 * 512, opB + (size_t)l * 512,
                    h, 512, nullptr, 0, MM, 512, 512, 0, stream);
    }
    // out = LN(h) with final gamma/beta
    k_ln<<<(MM + 3) / 4, 128, 0, stream>>>(h, fln_g, fln_b, (float*)d_out, 512);
}
